// AttentionBlock_20177756356775
// MI455X (gfx1250) — matrix operand order repaired and checked by reading, not compiled, not run on hardware
//
#include <hip/hip_runtime.h>
#include <stdint.h>
#include <stddef.h>

// ---------- types ----------
typedef __bf16 bf16_t;
typedef __attribute__((ext_vector_type(16))) bf16_t v16bf;
typedef __attribute__((ext_vector_type(8)))  bf16_t v8bf;
typedef __attribute__((ext_vector_type(8)))  float  v8f;

#define C_DIM   256
#define N_DIM   4096
#define HEADS   4
#define D_HEAD  64
#define GROUPS  16
#define QK_SCALE 0.35355339059327373f   // 64^(-1/4)

__device__ __forceinline__ bf16_t f2bf(float f) {
    union { float f; uint32_t u; } v; v.f = f;
    uint32_t r = (v.u + 0x7FFFu + ((v.u >> 16) & 1u)) >> 16;
    uint16_t h = (uint16_t)r;
    return __builtin_bit_cast(bf16_t, h);
}

__device__ __forceinline__ v8f wmma_bf16(v16bf a, v16bf b, v8f c) {
    return __builtin_amdgcn_wmma_f32_16x16x32_bf16(
        /*neg_a=*/false, a, /*neg_b=*/false, b,
        /*c_mod=*/(short)0, c, /*reuse_a=*/false, /*reuse_b=*/false);
}

// A-fragment: 16x32 (MxK), source row-major [m][k] with leading dim ld.
// lane<16: row m0+L, k = c0+0..15; lane>=16: row m0+L-16, k = c0+16..31
__device__ __forceinline__ v16bf load_a_frag(const bf16_t* A, int m0, int c0,
                                             int ld, int lane) {
    int r    = lane & 15;
    int koff = (lane >> 4) * 16;
    const bf16_t* p = A + (size_t)(m0 + r) * ld + c0 + koff;
    v8bf lo = *(const v8bf*)p;
    v8bf hi = *(const v8bf*)(p + 8);
    v16bf out;
#pragma unroll
    for (int i = 0; i < 8; i++) { out[i] = lo[i]; out[i + 8] = hi[i]; }
    return out;
}

// B-fragment: 32x16 (KxN), from *transposed* storage BT[n][k] (leading dim ld).
// lane<16: col n0+L, k = c0+0..15 (contiguous); lane>=16: col n0+L-16, k = c0+16..31
__device__ __forceinline__ v16bf load_b_frag(const bf16_t* BT, int n0, int c0,
                                             int ld, int lane) {
    int n    = lane & 15;
    int koff = (lane >> 4) * 16;
    const bf16_t* p = BT + (size_t)(n0 + n) * ld + c0 + koff;
    return *(const v16bf*)p;   // 32B contiguous, 32B aligned
}

// async global->LDS copy of 16 bytes (gfx1250 ASYNCcnt path)
__device__ __forceinline__ void async_copy_b128(uint32_t lds_off, const void* gptr) {
    uint64_t ga = (uint64_t)(uintptr_t)gptr;
    asm volatile("global_load_async_to_lds_b128 %0, %1, off"
                 :: "v"(lds_off), "v"(ga) : "memory");
}
__device__ __forceinline__ void wait_asynccnt0() {
    asm volatile("s_wait_asynccnt 0" ::: "memory");
}

// ---------- kernel 1: GroupNorm statistics ----------
__global__ void gn_stats_kernel(const float* __restrict__ x,
                                float* __restrict__ mu, float* __restrict__ rs) {
    __shared__ float red[256], red2[256];
    int g = blockIdx.x;
    const float* xp = x + (size_t)g * 16 * N_DIM;   // group is contiguous
    float s = 0.f, s2 = 0.f;
    for (int i = threadIdx.x; i < 16 * N_DIM; i += blockDim.x) {
        float v = xp[i]; s += v; s2 += v * v;
    }
    red[threadIdx.x] = s; red2[threadIdx.x] = s2;
    __syncthreads();
    for (int st = 128; st > 0; st >>= 1) {
        if ((int)threadIdx.x < st) {
            red[threadIdx.x]  += red[threadIdx.x + st];
            red2[threadIdx.x] += red2[threadIdx.x + st];
        }
        __syncthreads();
    }
    if (threadIdx.x == 0) {
        float inv = 1.f / (16.f * N_DIM);
        float mean = red[0] * inv;
        float var  = red2[0] * inv - mean * mean;
        mu[g] = mean;
        rs[g] = rsqrtf(var + 1e-5f);
    }
}

// ---------- kernel 2: normalize + transpose -> xn_t[n][c] (bf16) ----------
__global__ void gn_norm_t_kernel(const float* __restrict__ x,
                                 const float* __restrict__ gamma,
                                 const float* __restrict__ beta,
                                 const float* __restrict__ mu,
                                 const float* __restrict__ rs,
                                 bf16_t* __restrict__ xn_t) {
    __shared__ float tile[32][33];
    int c0 = blockIdx.x * 32, n0 = blockIdx.y * 32;
#pragma unroll
    for (int k = 0; k < 4; k++) {
        int cl = threadIdx.y + k * 8;
        tile[cl][threadIdx.x] = x[(size_t)(c0 + cl) * N_DIM + n0 + threadIdx.x];
    }
    __syncthreads();
#pragma unroll
    for (int k = 0; k < 4; k++) {
        int nl = threadIdx.y + k * 8;
        int cl = threadIdx.x;
        int c  = c0 + cl;
        int g  = c >> 4;
        float v = (tile[cl][nl] - mu[g]) * rs[g] * gamma[c] + beta[c];
        xn_t[(size_t)(n0 + nl) * C_DIM + c] = f2bf(v);
    }
}

// ---------- kernel 3: weights fp32 -> bf16 ----------
__global__ void wconv_kernel(const float* __restrict__ w_qkv,
                             const float* __restrict__ w_proj,
                             bf16_t* __restrict__ wq, bf16_t* __restrict__ wp) {
    int i = blockIdx.x * blockDim.x + threadIdx.x;
    const int NQ = 3 * C_DIM * C_DIM;           // 196608
    if (i < NQ) wq[i] = f2bf(w_qkv[i]);
    else {
        int j = i - NQ;
        if (j < C_DIM * C_DIM) wp[j] = f2bf(w_proj[j]);
    }
}

// ---------- kernel 4: QKV GEMM (768x4096, K=256), 16x64 strip per wave ----------
// q_t[h][n][r] (N x d), k_t[h][n][r] (N x d), v_[h][r][n] (d x N); q,k pre-scaled
__global__ void qkv_gemm_kernel(const bf16_t* __restrict__ wq,
                                const bf16_t* __restrict__ xn_t,
                                bf16_t* __restrict__ q_t,
                                bf16_t* __restrict__ k_t,
                                bf16_t* __restrict__ v_) {
    int lane = threadIdx.x & 31, wave = threadIdx.x >> 5;
    int n0 = blockIdx.x * 64;
    int o0 = (blockIdx.y * 4 + wave) * 16;
    v8f acc[4];
#pragma unroll
    for (int t = 0; t < 4; t++) acc[t] = (v8f){};
#pragma unroll
    for (int kk = 0; kk < 8; kk++) {
        v16bf a = load_a_frag(wq, o0, kk * 32, C_DIM, lane);   // reused 4x
#pragma unroll
        for (int t = 0; t < 4; t++) {
            v16bf b = load_b_frag(xn_t, n0 + t * 16, kk * 32, C_DIM, lane);
            acc[t] = wmma_bf16(a, b, acc[t]);
        }
    }
    int rhi = 8 * (lane >> 4);
#pragma unroll
    for (int t = 0; t < 4; t++) {
        int n = n0 + t * 16 + (lane & 15);
#pragma unroll
        for (int j = 0; j < 8; j++) {
            int o = o0 + j + rhi;
            int head = o / 192, r = o % 192;
            float val = acc[t][j];
            if (r < 128) val *= QK_SCALE;
            bf16_t hv = f2bf(val);
            if (r < 64)
                q_t[((size_t)(head * N_DIM + n)) * D_HEAD + r] = hv;
            else if (r < 128)
                k_t[((size_t)(head * N_DIM + n)) * D_HEAD + (r - 64)] = hv;
            else
                v_[((size_t)(head * D_HEAD + (r - 128))) * N_DIM + n] = hv;
        }
    }
}

// ---------- kernel 5: flash attention ----------
// Block = 4 waves = 64 query rows of one head. K/V chunk staged to LDS once per
// block via async global->LDS (ASYNCcnt), fragments then read with ds_load.
__global__ void flash_attn_kernel(const bf16_t* __restrict__ q_t,
                                  const bf16_t* __restrict__ k_t,
                                  const bf16_t* __restrict__ v_,
                                  bf16_t* __restrict__ h_t) {
    __shared__ bf16_t k_lds[32 * 64];        // [s_local][c]   4 KB
    __shared__ bf16_t v_lds[64 * 32];        // [c][s_local]   4 KB
    __shared__ bf16_t p_lds[4][16][32];      // per-wave P     4 KB
    int lane = threadIdx.x & 31, wave = threadIdx.x >> 5;
    int tid  = threadIdx.x;
    int head = blockIdx.x >> 6;              // 64 t-blocks per head
    int tblk = blockIdx.x & 63;
    int t0 = tblk * 64 + wave * 16;

    const bf16_t* qh = q_t + (size_t)head * N_DIM * D_HEAD;
    const char*   kg = (const char*)(k_t + (size_t)head * N_DIM * D_HEAD);
    const char*   vg = (const char*)(v_  + (size_t)head * D_HEAD * N_DIM);

    uint32_t k_base = (uint32_t)(uintptr_t)&k_lds[0];   // LDS byte offset
    uint32_t v_base = (uint32_t)(uintptr_t)&v_lds[0];

    // Q A-fragments are loop invariant (K=64 -> 2 chunks of 32)
    v16bf aq0 = load_a_frag(qh, t0, 0,  D_HEAD, lane);
    v16bf aq1 = load_a_frag(qh, t0, 32, D_HEAD, lane);

    float m[8], l[8];
    v8f acc[4];
#pragma unroll
    for (int j = 0; j < 8; j++) { m[j] = -1e30f; l[j] = 0.f; }
#pragma unroll
    for (int t = 0; t < 4; t++) acc[t] = (v8f){};

    for (int s0 = 0; s0 < N_DIM; s0 += 32) {
        // ---- stage K chunk (32x64, contiguous 4KB) and V chunk (64 x 64B rows)
#pragma unroll
        for (int rnd = 0; rnd < 2; rnd++) {
            int off = (tid + rnd * 128) * 16;           // 0..4080
            async_copy_b128(k_base + off, kg + (size_t)s0 * 128 + off);
            int c = off >> 6, within = off & 63;        // v_lds[c][*] row = 64B
            async_copy_b128(v_base + off,
                            vg + (size_t)c * (N_DIM * 2) + (size_t)s0 * 2 + within);
        }
        wait_asynccnt0();
        __syncthreads();
        if (s0 + 32 < N_DIM) {                          // warm next K chunk
            __builtin_prefetch(kg + (size_t)(s0 + 32) * 128, 0, 0);
            __builtin_prefetch(vg + (size_t)(s0 + 32) * 2, 0, 0);
        }

        // ---- S tile: 16 t-rows x 32 s-cols (two 16x16 accumulators), from LDS
        v8f sa0 = {}, sa1 = {};
        {
            v16bf b00 = load_b_frag(k_lds, 0,  0,  D_HEAD, lane);
            v16bf b01 = load_b_frag(k_lds, 16, 0,  D_HEAD, lane);
            sa0 = wmma_bf16(aq0, b00, sa0);
            sa1 = wmma_bf16(aq0, b01, sa1);
            v16bf b10 = load_b_frag(k_lds, 0,  32, D_HEAD, lane);
            v16bf b11 = load_b_frag(k_lds, 16, 32, D_HEAD, lane);
            sa0 = wmma_bf16(aq1, b10, sa0);
            sa1 = wmma_bf16(aq1, b11, sa1);
        }
        // ---- online softmax: C/D layout -> lanes 0-15 rows j, lanes 16-31 rows
        // j+8; xor masks 1..8 reduce both halves independently & simultaneously.
        int cI = lane & 15;
#pragma unroll
        for (int j = 0; j < 8; j++) {
            float cmax = fmaxf(sa0[j], sa1[j]);
#pragma unroll
            for (int dd = 1; dd < 16; dd <<= 1) cmax = fmaxf(cmax, __shfl_xor(cmax, dd));
            float mn = fmaxf(m[j], cmax);
            float al = __expf(m[j] - mn);
            float p0 = __expf(sa0[j] - mn);
            float p1 = __expf(sa1[j] - mn);
            float rsum = p0 + p1;
#pragma unroll
            for (int dd = 1; dd < 16; dd <<= 1) rsum += __shfl_xor(rsum, dd);
            m[j] = mn;
            l[j] = l[j] * al + rsum;
#pragma unroll
            for (int t = 0; t < 4; t++) acc[t][j] *= al;
            int r = j + 8 * (lane >> 4);
            p_lds[wave][r][cI]      = f2bf(p0);
            p_lds[wave][r][cI + 16] = f2bf(p1);
        }
        // ---- O += P @ V  (P as A-fragment from LDS, V fragments from LDS)
        v16bf ap  = load_a_frag(&p_lds[wave][0][0], 0, 0, 32, lane);
        v16bf bv0 = load_b_frag(v_lds, 0,  0, 32, lane);
        v16bf bv1 = load_b_frag(v_lds, 16, 0, 32, lane);
        v16bf bv2 = load_b_frag(v_lds, 32, 0, 32, lane);
        v16bf bv3 = load_b_frag(v_lds, 48, 0, 32, lane);
        acc[0] = wmma_bf16(ap, bv0, acc[0]);
        acc[1] = wmma_bf16(ap, bv1, acc[1]);
        acc[2] = wmma_bf16(ap, bv2, acc[2]);
        acc[3] = wmma_bf16(ap, bv3, acc[3]);
        __syncthreads();   // before next chunk overwrites k_lds/v_lds
    }
    // ---- epilogue: O /= l, store h_t[n][channel] (N x C, bf16)
    int cI = lane & 15, rhi = 8 * (lane >> 4);
#pragma unroll
    for (int j = 0; j < 8; j++) {
        float linv = 1.f / l[j];
        int tg = t0 + j + rhi;
#pragma unroll
        for (int t = 0; t < 4; t++) {
            int ch = head * D_HEAD + t * 16 + cI;
            h_t[(size_t)tg * C_DIM + ch] = f2bf(acc[t][j] * linv);
        }
    }
}

// ---------- kernel 6: proj GEMM + bias + residual, 16x64 strip per wave ----------
__global__ void proj_gemm_kernel(const bf16_t* __restrict__ wp,
                                 const bf16_t* __restrict__ h_t,
                                 const float* __restrict__ b_proj,
                                 const float* __restrict__ x,
                                 float* __restrict__ out) {
    int lane = threadIdx.x & 31, wave = threadIdx.x >> 5;
    int n0 = blockIdx.x * 64;
    int o0 = (blockIdx.y * 4 + wave) * 16;
    v8f acc[4];
#pragma unroll
    for (int t = 0; t < 4; t++) acc[t] = (v8f){};
#pragma unroll
    for (int kk = 0; kk < 8; kk++) {
        v16bf a = load_a_frag(wp, o0, kk * 32, C_DIM, lane);   // reused 4x
#pragma unroll
        for (int t = 0; t < 4; t++) {
            v16bf b = load_b_frag(h_t, n0 + t * 16, kk * 32, C_DIM, lane);
            acc[t] = wmma_bf16(a, b, acc[t]);
        }
    }
    int rhi = 8 * (lane >> 4);
#pragma unroll
    for (int t = 0; t < 4; t++) {
        int n = n0 + t * 16 + (lane & 15);
#pragma unroll
        for (int j = 0; j < 8; j++) {
            int o = o0 + j + rhi;
            size_t idx = (size_t)o * N_DIM + n;
            out[idx] = acc[t][j] + b_proj[o] + x[idx];
        }
    }
}

// ---------- host launcher ----------
extern "C" void kernel_launch(void* const* d_in, const int* in_sizes, int n_in,
                              void* d_out, int out_size, void* d_ws, size_t ws_size,
                              hipStream_t stream) {
    const float* x      = (const float*)d_in[0];
    const float* gamma  = (const float*)d_in[1];
    const float* beta   = (const float*)d_in[2];
    const float* w_qkv  = (const float*)d_in[3];
    const float* w_proj = (const float*)d_in[4];
    const float* b_proj = (const float*)d_in[5];
    float* out = (float*)d_out;

    char* ws = (char*)d_ws;
    size_t off = 0;
    auto alloc = [&](size_t bytes) -> char* {
        char* p = ws + off;
        off += (bytes + 255) & ~(size_t)255;
        return p;
    };
    float*  mu   = (float*)alloc(GROUPS * sizeof(float));
    float*  rs   = (float*)alloc(GROUPS * sizeof(float));
    bf16_t* wq   = (bf16_t*)alloc((size_t)3 * C_DIM * C_DIM * 2);
    bf16_t* wp   = (bf16_t*)alloc((size_t)C_DIM * C_DIM * 2);
    bf16_t* xn_t = (bf16_t*)alloc((size_t)N_DIM * C_DIM * 2);
    bf16_t* q_t  = (bf16_t*)alloc((size_t)HEADS * N_DIM * D_HEAD * 2);
    bf16_t* k_t  = (bf16_t*)alloc((size_t)HEADS * N_DIM * D_HEAD * 2);
    bf16_t* v_   = (bf16_t*)alloc((size_t)HEADS * D_HEAD * N_DIM * 2);
    bf16_t* h_t  = (bf16_t*)alloc((size_t)N_DIM * C_DIM * 2);

    gn_stats_kernel<<<GROUPS, 256, 0, stream>>>(x, mu, rs);
    wconv_kernel<<<(3 * C_DIM * C_DIM + C_DIM * C_DIM) / 256, 256, 0, stream>>>(
        w_qkv, w_proj, wq, wp);
    gn_norm_t_kernel<<<dim3(C_DIM / 32, N_DIM / 32), dim3(32, 8), 0, stream>>>(
        x, gamma, beta, mu, rs, xn_t);
    qkv_gemm_kernel<<<dim3(N_DIM / 64, (3 * C_DIM) / 64), 128, 0, stream>>>(
        wq, xn_t, q_t, k_t, v_);
    flash_attn_kernel<<<HEADS * (N_DIM / 64), 128, 0, stream>>>(q_t, k_t, v_, h_t);
    proj_gemm_kernel<<<dim3(N_DIM / 64, C_DIM / 64), 128, 0, stream>>>(
        wp, h_t, b_proj, x, out);
}
